// MultiViewCrossAttention_5265629905036
// MI455X (gfx1250) — matrix operand order repaired and checked by reading, not compiled, not run on hardware
//
#include <hip/hip_runtime.h>

// ---------------------------------------------------------------------------
// MultiViewCrossAttention for MI455X (gfx1250), bf16 WMMA pipeline.
// B=4, V=8, N=256, C=768, H=12, DH=64.
// V is stored transposed per (b,h) as vt[d][m] so the P@V B-fragments are
// single contiguous 32-byte loads (same pattern as the Q@K^T phase).
// ---------------------------------------------------------------------------

#define Bq 4
#define Vv 8
#define Nn 256
#define Cc 768
#define Hh 12
#define DH 64
#define MROWS (Bq * Vv * Nn)      // 8192
#define KEYS (Vv * Nn)            // 2048 keys per (b,h)
#define C3 (3 * Cc)               // 2304

typedef unsigned short u16;
typedef __attribute__((ext_vector_type(16))) u16     v16u;
typedef __attribute__((ext_vector_type(8)))  u16     v8u;
typedef __attribute__((ext_vector_type(16))) __bf16  v16bf;
typedef __attribute__((ext_vector_type(8)))  float   v8f;

__device__ __forceinline__ u16 f32_to_bf16_bits(float f) {
  unsigned u = __builtin_bit_cast(unsigned, f);
  u += 0x7FFFu + ((u >> 16) & 1u);              // round-to-nearest-even
  return (u16)(u >> 16);
}

// ---------------------------------------------------------------------------
// Stage 0: f32 -> bf16 conversion
// ---------------------------------------------------------------------------
__global__ void cvt_bf16_kernel(const float* __restrict__ src,
                                u16* __restrict__ dst, int n) {
  int i = blockIdx.x * blockDim.x + threadIdx.x;
  if (i < n) dst[i] = f32_to_bf16_bits(src[i]);
}

// ---------------------------------------------------------------------------
// Stage 1: QKV projection GEMM.  qkv = feats @ w_qkv^T + b_qkv
//   A: feats bf16  [8192][768]
//   B: w_qkv bf16  [2304][768]   (B(k,n) = W[n][k])
// Block = 256 threads = 8 waves; wave tile 16x16; block tile 64 x 32.
// Epilogue scatters q (scaled 1/8) [b][h][v][n][d], k [b][h][m][d],
// and v TRANSPOSED [b][h][d][m].
// ---------------------------------------------------------------------------
__global__ void qkv_gemm_kernel(const u16* __restrict__ A,
                                const u16* __restrict__ Wb,
                                const float* __restrict__ bias,
                                u16* __restrict__ qw,
                                u16* __restrict__ kw,
                                u16* __restrict__ vt) {
  const int tid  = threadIdx.x;
  const int lane = tid & 31;
  const int wave = tid >> 5;
  const int half = lane >> 4;       // 0: lanes 0-15, 1: lanes 16-31
  const int l15  = lane & 15;
  const int wm = wave >> 1;         // 0..3
  const int wn = wave & 1;          // 0..1
  const int row0 = blockIdx.y * 64 + wm * 16;
  const int col0 = blockIdx.x * 32 + wn * 16;

  const u16* arow = A  + (size_t)(row0 + l15) * Cc;
  const u16* brow = Wb + (size_t)(col0 + l15) * Cc;
  const int akb = half * 16;        // A-fragment K base (contiguous k = 16*half + s)
  const int bkb = half * 16;        // B-fragment K base

  v8f acc = {};
#pragma unroll 4
  for (int ks = 0; ks < Cc / 32; ++ks) {
    const int k0 = ks * 32;
    // A fragment: two contiguous 8-elem runs: [akb..akb+7], [akb+8..akb+15]
    v8u r0 = *(const v8u*)(arow + k0 + akb);
    v8u r1 = *(const v8u*)(arow + k0 + akb + 8);
    v16u au;
#pragma unroll
    for (int i = 0; i < 8; ++i) { au[i] = r0[i]; au[8 + i] = r1[i]; }
    v16bf af = __builtin_bit_cast(v16bf, au);
    // B fragment: contiguous 16 elems along K
    v16u bu = *(const v16u*)(brow + k0 + bkb);
    v16bf bf = __builtin_bit_cast(v16bf, bu);
    acc = __builtin_amdgcn_wmma_f32_16x16x32_bf16(false, af, false, bf,
                                                  (short)0, acc, false, false);
  }

  const int col = col0 + l15;
  const float bv = bias[col];
#pragma unroll
  for (int r = 0; r < 8; ++r) {
    const int row = row0 + r + half * 8;      // D layout: lanes16-31 hold M=r+8
    float val = acc[r] + bv;
    const int b_ = row >> 11;                 // / (V*N)
    const int v_ = (row >> 8) & 7;
    const int n_ = row & 255;
    if (col < Cc) {                           // Q (scaled by 1/sqrt(DH))
      const int h_ = col >> 6, d_ = col & 63;
      qw[((((size_t)b_ * Hh + h_) * Vv + v_) * Nn + n_) * DH + d_] =
          f32_to_bf16_bits(val * 0.125f);
    } else if (col < 2 * Cc) {                // K: [b][h][m][d]
      const int j = col - Cc, h_ = j >> 6, d_ = j & 63;
      kw[(((size_t)b_ * Hh + h_) * KEYS + (v_ * Nn + n_)) * DH + d_] =
          f32_to_bf16_bits(val);
    } else {                                  // V transposed: [b][h][d][m]
      const int j = col - 2 * Cc, h_ = j >> 6, d_ = j & 63;
      vt[(((size_t)b_ * Hh + h_) * DH + d_) * KEYS + (v_ * Nn + n_)] =
          f32_to_bf16_bits(val);
    }
  }
}

// ---------------------------------------------------------------------------
// Stage 2: attention. One block per (b, h, view, 32-row q chunk).
// 8 waves == 8 key views: wave w computes scores vs key-view w (or fills NEG
// if masked).  Scores 32x2048 f32 in dynamic LDS (256 KB), 3-sweep softmax,
// then O = P @ V via WMMA with on-the-fly bf16 packing (V read from the
// transposed layout -> contiguous 32B B-fragment loads).
// ---------------------------------------------------------------------------
__global__ void attn_kernel(const u16* __restrict__ qw,
                            const u16* __restrict__ kw,
                            const u16* __restrict__ vt,
                            const unsigned char* __restrict__ is_ref,
                            u16* __restrict__ ctxw) {
  extern __shared__ char smem_raw[];
  float* S   = (float*)smem_raw;                               // 32 x 2048
  float* red = (float*)(smem_raw + (size_t)32 * KEYS * 4);     // 32 x 8

  const int bid   = blockIdx.x;
  const int chunk = bid & 7;
  const int v_    = (bid >> 3) & 7;
  const int h_    = (bid >> 6) % Hh;
  const int b_    = bid / (8 * 8 * Hh);

  const int tid  = threadIdx.x;
  const int lane = tid & 31;
  const int wave = tid >> 5;            // == key view index
  const int half = lane >> 4;
  const int l15  = lane & 15;

  const bool myref = is_ref[b_ * Vv + v_] != 0;
  const bool kvref = is_ref[b_ * Vv + wave] != 0;
  const bool allowedView = (myref != kvref);

  const size_t qbase  = ((((size_t)b_ * Hh + h_) * Vv + v_) * Nn + chunk * 32) * DH;
  const size_t kvbase = ((size_t)b_ * Hh + h_) * KEYS * DH;   // K and Vt share extent
  const int akb = half * 16;

  // Preload 4 Q A-fragments (2 row-tiles x 2 K-steps over DH=64)
  v16bf qfrag[2][2];
#pragma unroll
  for (int mt = 0; mt < 2; ++mt) {
    const u16* qrow = qw + qbase + (size_t)(mt * 16 + l15) * DH;
#pragma unroll
    for (int ks = 0; ks < 2; ++ks) {
      v8u r0 = *(const v8u*)(qrow + ks * 32 + akb);
      v8u r1 = *(const v8u*)(qrow + ks * 32 + akb + 8);
      v16u au;
#pragma unroll
      for (int i = 0; i < 8; ++i) { au[i] = r0[i]; au[8 + i] = r1[i]; }
      qfrag[mt][ks] = __builtin_bit_cast(v16bf, au);
    }
  }

  // ---- Phase 1: scores ----
  if (allowedView) {
    for (int ct = 0; ct < 16; ++ct) {                   // 16 col-tiles = 256 keys
      const int colbase = wave * 256 + ct * 16;
      const u16* krow = kw + kvbase + (size_t)(colbase + l15) * DH;
      v16bf bfr[2];
#pragma unroll
      for (int ks = 0; ks < 2; ++ks) {
        v16u bu = *(const v16u*)(krow + ks * 32 + half * 16);
        bfr[ks] = __builtin_bit_cast(v16bf, bu);
      }
#pragma unroll
      for (int mt = 0; mt < 2; ++mt) {
        v8f acc = {};
        acc = __builtin_amdgcn_wmma_f32_16x16x32_bf16(false, qfrag[mt][0], false,
                                                      bfr[0], (short)0, acc, false, false);
        acc = __builtin_amdgcn_wmma_f32_16x16x32_bf16(false, qfrag[mt][1], false,
                                                      bfr[1], (short)0, acc, false, false);
#pragma unroll
        for (int r = 0; r < 8; ++r) {
          const int row = mt * 16 + r + half * 8;
          S[row * KEYS + colbase + l15] = acc[r];
        }
      }
    }
  } else {
    for (int i = lane; i < 32 * 256; i += 32) {
      S[(i >> 8) * KEYS + wave * 256 + (i & 255)] = -1e9f;
    }
  }
  __syncthreads();

  // ---- Phase 2: softmax over 2048 keys, 8 threads per row ----
  {
    const int row = tid >> 3;
    const int seg = tid & 7;
    float* Srow = S + row * KEYS + seg * 256;
    float m = -3.4e38f;
    for (int i = 0; i < 256; ++i) m = fmaxf(m, Srow[i]);
    red[row * 8 + seg] = m;
    __syncthreads();
    float rmax = red[row * 8];
#pragma unroll
    for (int i = 1; i < 8; ++i) rmax = fmaxf(rmax, red[row * 8 + i]);
    float psum = 0.f;
    for (int i = 0; i < 256; ++i) {
      float e = __expf(Srow[i] - rmax);
      Srow[i] = e;
      psum += e;
    }
    __syncthreads();
    red[row * 8 + seg] = psum;
    __syncthreads();
    float rsum = 0.f;
#pragma unroll
    for (int i = 0; i < 8; ++i) rsum += red[row * 8 + i];
    const float inv = 1.0f / rsum;
    for (int i = 0; i < 256; ++i) Srow[i] *= inv;
  }
  __syncthreads();

  // ---- Phase 3: O = P @ V ; 8 output tiles (2 row-tiles x 4 d-tiles) ----
  {
    const int mt = wave >> 2;
    const int nt = wave & 3;
    v8f acc = {};
    const float* srow = S + (mt * 16 + l15) * KEYS;
    // B-fragment source: vt row for this lane's d column, contiguous over m.
    const u16* vrow = vt + kvbase + (size_t)(nt * 16 + l15) * KEYS;
    for (int ks = 0; ks < KEYS / 32; ++ks) {
      const float* sp = srow + ks * 32 + akb;
      v16u au;
#pragma unroll
      for (int i = 0; i < 8; ++i) au[i]     = f32_to_bf16_bits(sp[i]);
#pragma unroll
      for (int i = 0; i < 8; ++i) au[8 + i] = f32_to_bf16_bits(sp[8 + i]);
      v16bf af = __builtin_bit_cast(v16bf, au);
      v16u bu = *(const v16u*)(vrow + ks * 32 + half * 16);
      v16bf bf = __builtin_bit_cast(v16bf, bu);
      acc = __builtin_amdgcn_wmma_f32_16x16x32_bf16(false, af, false, bf,
                                                    (short)0, acc, false, false);
    }
#pragma unroll
    for (int r = 0; r < 8; ++r) {
      const int n_ = chunk * 32 + mt * 16 + r + half * 8;
      const int c_ = h_ * DH + nt * 16 + l15;
      ctxw[(((size_t)b_ * Vv + v_) * Nn + n_) * Cc + c_] = f32_to_bf16_bits(acc[r]);
    }
  }
}

// ---------------------------------------------------------------------------
// Stage 3: out projection + bias + pass-through select.
// ---------------------------------------------------------------------------
__global__ void out_gemm_kernel(const u16* __restrict__ A,     // ctx bf16 [8192][768]
                                const u16* __restrict__ Wb,    // w_out bf16 [768][768]
                                const float* __restrict__ bias,
                                const float* __restrict__ feats,
                                const unsigned char* __restrict__ is_ref,
                                float* __restrict__ out) {
  const int tid  = threadIdx.x;
  const int lane = tid & 31;
  const int wave = tid >> 5;
  const int half = lane >> 4;
  const int l15  = lane & 15;
  const int wm = wave >> 1;
  const int wn = wave & 1;
  const int row0 = blockIdx.y * 64 + wm * 16;
  const int col0 = blockIdx.x * 32 + wn * 16;

  const u16* arow = A  + (size_t)(row0 + l15) * Cc;
  const u16* brow = Wb + (size_t)(col0 + l15) * Cc;
  const int akb = half * 16;
  const int bkb = half * 16;

  v8f acc = {};
#pragma unroll 4
  for (int ks = 0; ks < Cc / 32; ++ks) {
    const int k0 = ks * 32;
    v8u r0 = *(const v8u*)(arow + k0 + akb);
    v8u r1 = *(const v8u*)(arow + k0 + akb + 8);
    v16u au;
#pragma unroll
    for (int i = 0; i < 8; ++i) { au[i] = r0[i]; au[8 + i] = r1[i]; }
    v16bf af = __builtin_bit_cast(v16bf, au);
    v16u bu = *(const v16u*)(brow + k0 + bkb);
    v16bf bf = __builtin_bit_cast(v16bf, bu);
    acc = __builtin_amdgcn_wmma_f32_16x16x32_bf16(false, af, false, bf,
                                                  (short)0, acc, false, false);
  }

  const int col = col0 + l15;
  const float bv = bias[col];
#pragma unroll
  for (int r = 0; r < 8; ++r) {
    const int row = row0 + r + half * 8;
    const int b_ = row >> 11;
    const int v_ = (row >> 8) & 7;
    const bool myref = is_ref[b_ * Vv + v_] != 0;
    bool has = false;
#pragma unroll
    for (int w = 0; w < Vv; ++w) has = has || ((is_ref[b_ * Vv + w] != 0) != myref);
    const size_t idx = (size_t)row * Cc + col;
    out[idx] = has ? (acc[r] + bv) : feats[idx];
  }
}

// ---------------------------------------------------------------------------
// Launch
// ---------------------------------------------------------------------------
extern "C" void kernel_launch(void* const* d_in, const int* in_sizes, int n_in,
                              void* d_out, int out_size, void* d_ws, size_t ws_size,
                              hipStream_t stream) {
  const float*         feats  = (const float*)d_in[0];
  const unsigned char* is_ref = (const unsigned char*)d_in[1];   // jnp.bool_ = 1 byte
  const float*         w_qkv  = (const float*)d_in[2];
  const float*         b_qkv  = (const float*)d_in[3];
  const float*         w_out  = (const float*)d_in[4];
  const float*         b_out  = (const float*)d_in[5];
  float* out = (float*)d_out;

  char* ws = (char*)d_ws;
  size_t off = 0;
  u16* featsb = (u16*)(ws + off); off += (size_t)MROWS * Cc * 2;        // 12.6 MB
  u16* wqkvb  = (u16*)(ws + off); off += (size_t)C3 * Cc * 2;           //  3.5 MB
  u16* woutb  = (u16*)(ws + off); off += (size_t)Cc * Cc * 2;           //  1.2 MB
  u16* qw     = (u16*)(ws + off); off += (size_t)Bq * Hh * KEYS * DH * 2;
  u16* kw     = (u16*)(ws + off); off += (size_t)Bq * Hh * KEYS * DH * 2;
  u16* vt     = (u16*)(ws + off); off += (size_t)Bq * Hh * KEYS * DH * 2;
  u16* ctxb   = featsb;  // feats bf16 dead after stage 1; reuse for ctx

  // Stage 0: conversions
  {
    const int nf = MROWS * Cc, nq = C3 * Cc, no = Cc * Cc;
    cvt_bf16_kernel<<<(nf + 255) / 256, 256, 0, stream>>>(feats, featsb, nf);
    cvt_bf16_kernel<<<(nq + 255) / 256, 256, 0, stream>>>(w_qkv, wqkvb, nq);
    cvt_bf16_kernel<<<(no + 255) / 256, 256, 0, stream>>>(w_out, woutb, no);
  }

  // Stage 1: QKV projection (grid: 2304/32 x 8192/64)
  qkv_gemm_kernel<<<dim3(C3 / 32, MROWS / 64), 256, 0, stream>>>(
      featsb, wqkvb, b_qkv, qw, kw, vt);

  // Stage 2: attention (B*H*V*(N/32) blocks, 256 KB + 1 KB dynamic LDS)
  {
    const size_t lds = (size_t)32 * KEYS * sizeof(float) + 32 * 8 * sizeof(float);
    attn_kernel<<<Bq * Hh * Vv * (Nn / 32), 256, lds, stream>>>(
        qw, kw, vt, is_ref, ctxb);
  }

  // Stage 3: out projection + select (grid: 768/32 x 8192/64)
  out_gemm_kernel<<<dim3(Cc / 32, MROWS / 64), 256, 0, stream>>>(
      ctxb, woutb, b_out, feats, is_ref, out);
}
